// MoE_27255862461164
// MI455X (gfx1250) — matrix operand order repaired and checked by reading, compiled, not run on hardware
//
#include <hip/hip_runtime.h>

typedef __attribute__((ext_vector_type(16))) _Float16 v16h;
typedef __attribute__((ext_vector_type(8)))  float    v8f;

namespace {
constexpr int   N_PTS   = 262144;
constexpr int   NEXP    = 8;
constexpr int   WAVES   = 16;          // 512 threads / block
constexpr int   BLOCKS  = 256;
constexpr int   SP      = 66;          // padded pitch in halves (33 dwords -> conflict free)
constexpr int   MATSZ   = 64 * SP;     // halves per 64x64 matrix
constexpr int   NMAT    = 25;          // gw1 + 8*we0 + 8*wm0 + 8*wm1
constexpr float BWD     = 45.0f;
constexpr float LN_EPS  = 1e-5f;

// float scratch layout (indices in floats)
constexpr int F_FW4 = 0;               // fw padded [64][4]
constexpr int F_FB  = F_FW4 + 256;     // 64
constexpr int F_GB1 = F_FB  + 64;
constexpr int F_LNG = F_GB1 + 64;
constexpr int F_LNB = F_LNG + 64;
constexpr int F_GW2 = F_LNB + 64;      // 8*64
constexpr int F_GB2 = F_GW2 + 512;     // 8
constexpr int F_BE0 = F_GB2 + 8;       // 8*64
constexpr int F_BM  = F_BE0 + 512;     // 2*8*64
constexpr int F_WL  = F_BM  + 1024;    // 8*64
constexpr int F_BL  = F_WL  + 512;     // 8
constexpr int F_TOT = F_BL  + 8;

constexpr size_t SM_W  = size_t(NMAT) * MATSZ * 2;      // 211200 B
constexpr size_t SM_ST = size_t(WAVES) * 16 * SP * 2;   //  33792 B
constexpr size_t SM_F  = size_t(F_TOT) * 4;             //  12352 B
constexpr size_t SM_M  = size_t(WAVES) * 16 * 4;        //   1024 B
constexpr size_t SMEM_BYTES = SM_W + SM_ST + SM_F + SM_M; // ~258 KB < 320 KB/WGP
}

union AFrag { v16h v; _Float16 h[16]; unsigned u[8]; };

// B fragment (32x16) for col-tile t, K-block kb.
// b[i] = W[col][32*kb + (i<8 ? 8*half+i : 16+8*half+(i-8))]  (W row-major 64 x SP halves in LDS)
__device__ __forceinline__ v16h loadB(const _Float16* __restrict__ W,
                                      int t, int kb, int l16, int half) {
  const unsigned* rowp = (const unsigned*)(W + (16 * t + l16) * SP);
  const int base = 16 * kb + 4 * half;   // dword offset in row
  union { v16h v; unsigned u[8]; } b;
#pragma unroll
  for (int j = 0; j < 8; ++j) b.u[j] = rowp[base + j + (j & 4)];
  return b.v;
}

__device__ __forceinline__ void zeroAcc(v8f* a) {
#pragma unroll
  for (int t = 0; t < 4; ++t)
    a[t] = (v8f){0.f, 0.f, 0.f, 0.f, 0.f, 0.f, 0.f, 0.f};
}

// D[16x64] += A[16x64] * W^T  (8 x V_WMMA_F32_16X16X32_F16)
// All 8 B fragments are preloaded into distinct registers first so the 32
// ds_load_2addr_b32 pipeline, then the 8 WMMAs issue back-to-back with no
// WAR hazards (no v_nop bursts) and only partial dscnt waits.
__device__ __forceinline__ void gemm64(const AFrag* a, const _Float16* __restrict__ W,
                                       v8f* acc, int l16, int half) {
  v16h bf[8];
#pragma unroll
  for (int t = 0; t < 4; ++t)
#pragma unroll
    for (int kb = 0; kb < 2; ++kb)
      bf[t * 2 + kb] = loadB(W, t, kb, l16, half);
#pragma unroll
  for (int t = 0; t < 4; ++t) {
    acc[t] = __builtin_amdgcn_wmma_f32_16x16x32_f16(
        false, a[0].v, false, bf[t * 2 + 0], (short)0, acc[t], false, false);
    acc[t] = __builtin_amdgcn_wmma_f32_16x16x32_f16(
        false, a[1].v, false, bf[t * 2 + 1], (short)0, acc[t], false, false);
  }
}

// h = sin(w0 * (acc + bias[col])) -> stage (f16, row-major, pitch SP)
__device__ __forceinline__ void actStage(const v8f* acc, const float* __restrict__ bias,
                                         float w0, _Float16* __restrict__ stw,
                                         int l16, int half) {
#pragma unroll
  for (int t = 0; t < 4; ++t) {
    const int col = 16 * t + l16;
    const float bc = bias[col];
#pragma unroll
    for (int j = 0; j < 8; ++j) {
      float v = __sinf(w0 * (acc[t][j] + bc));
      stw[(j + 8 * half) * SP + col] = (_Float16)v;
    }
  }
}

// regather staged 16x64 f16 tile into A fragments (dword-packed, conflict-free)
__device__ __forceinline__ void reloadA(const _Float16* __restrict__ stw,
                                        AFrag* a, int l16, int half) {
  const unsigned* rp = (const unsigned*)(stw + l16 * SP);
#pragma unroll
  for (int kb = 0; kb < 2; ++kb) {
    const int base = 16 * kb + 4 * half;
#pragma unroll
    for (int j = 0; j < 4; ++j) a[kb].u[j] = rp[base + j];
#pragma unroll
    for (int j = 0; j < 4; ++j) a[kb].u[4 + j] = rp[base + 8 + j];
  }
}

__global__ void __launch_bounds__(WAVES * 32)
moe_kernel(const float* __restrict__ coords, const float* __restrict__ fw,
           const float* __restrict__ fb,     const float* __restrict__ gw1,
           const float* __restrict__ gb1,    const float* __restrict__ lng,
           const float* __restrict__ lnb,    const float* __restrict__ gw2,
           const float* __restrict__ gb2,    const float* __restrict__ we0,
           const float* __restrict__ be0,    const float* __restrict__ bm,
           const float* __restrict__ wm,     const float* __restrict__ wl,
           const float* __restrict__ bl,     float* __restrict__ out,
           float* __restrict__ counts)
{
  extern __shared__ char smem[];
  _Float16* Wlds  = (_Float16*)smem;
  _Float16* stage = (_Float16*)(smem + SM_W);
  float*    fsm   = (float*)(smem + SM_W + SM_ST);
  unsigned* mlds  = (unsigned*)(smem + SM_W + SM_ST + SM_F);

  const int tid = threadIdx.x;

  // ---- one-time cooperative preload: all weights -> f16 LDS ----
  for (int idx = tid; idx < NMAT * 4096; idx += WAVES * 32) {
    const int m = idx >> 12, r = (idx >> 6) & 63, c = idx & 63;
    const float* src = (m == 0) ? gw1 : (m < 9) ? (we0 + (m - 1) * 4096)
                                                : (wm + (m - 9) * 4096);
    Wlds[m * MATSZ + r * SP + c] = (_Float16)src[r * 64 + c];
  }
  for (int i = tid; i < 64 * 3; i += WAVES * 32)
    fsm[F_FW4 + (i / 3) * 4 + (i % 3)] = fw[i];
  for (int i = tid; i < 64; i += WAVES * 32) {
    fsm[F_FB + i] = fb[i]; fsm[F_GB1 + i] = gb1[i];
    fsm[F_LNG + i] = lng[i]; fsm[F_LNB + i] = lnb[i];
  }
  for (int i = tid; i < 512; i += WAVES * 32) {
    fsm[F_GW2 + i] = gw2[i]; fsm[F_BE0 + i] = be0[i]; fsm[F_WL + i] = wl[i];
  }
  for (int i = tid; i < 1024; i += WAVES * 32) fsm[F_BM + i] = bm[i];
  if (tid < 8) { fsm[F_GB2 + tid] = gb2[tid]; fsm[F_BL + tid] = bl[tid]; }
  __syncthreads();

  const int lane = tid & 31, half = lane >> 4, l16 = lane & 15;
  const int wslot = tid >> 5;
  _Float16* stw = stage + wslot * 16 * SP;
  unsigned* msw = mlds + wslot * 16;

  float cntAcc = 0.f;
  const int wid = blockIdx.x * WAVES + wslot;
  const int tilesTotal = N_PTS / 16;

  for (int tile = wid; tile < tilesTotal; tile += BLOCKS * WAVES) {
    const int row = tile * 16 + l16;
    const float c0 = coords[row * 3], c1 = coords[row * 3 + 1], c2 = coords[row * 3 + 2];

    // prefetch next tile's coords (global_prefetch_b8); uniform branch per wave
    {
      const int nt = tile + BLOCKS * WAVES;
      if (nt < tilesTotal)
        __builtin_prefetch(coords + (nt * 16 + l16) * 3, 0, 1);
    }

    // feature = coords @ fw.T + fb, built directly into A-fragment layout
    AFrag af[2];
#pragma unroll
    for (int kb = 0; kb < 2; ++kb) {
#pragma unroll
      for (int i = 0; i < 16; ++i) {
        const int k = 32 * kb + ((i < 8) ? (8 * half + i) : (16 + 8 * half + (i - 8)));
        float v = fsm[F_FB + k] + c0 * fsm[F_FW4 + 4 * k]
                                + c1 * fsm[F_FW4 + 4 * k + 1]
                                + c2 * fsm[F_FW4 + 4 * k + 2];
        af[kb].h[i] = (_Float16)v;
      }
    }

    // ---- gate: Linear(64,64) -> LayerNorm -> Linear(64,8) -> top2 on logits ----
    v8f acc[4];
    zeroAcc(acc);
    gemm64(af, Wlds /* m=0: gw1 */, acc, l16, half);
#pragma unroll
    for (int t = 0; t < 4; ++t) {
      const float bc = fsm[F_GB1 + 16 * t + l16];
#pragma unroll
      for (int j = 0; j < 8; ++j) acc[t][j] += bc;
    }

    float mu[8], rs[8];
#pragma unroll
    for (int j = 0; j < 8; ++j) {
      float s = acc[0][j] + acc[1][j] + acc[2][j] + acc[3][j];
      float q = acc[0][j] * acc[0][j] + acc[1][j] * acc[1][j]
              + acc[2][j] * acc[2][j] + acc[3][j] * acc[3][j];
#pragma unroll
      for (int msk = 1; msk < 16; msk <<= 1) {
        s += __shfl_xor(s, msk, 32);
        q += __shfl_xor(q, msk, 32);
      }
      const float mean = s * (1.f / 64.f);
      mu[j] = mean;
      rs[j] = rsqrtf(q * (1.f / 64.f) - mean * mean + LN_EPS);
    }
#pragma unroll
    for (int t = 0; t < 4; ++t) {
      const int col = 16 * t + l16;
      const float gg = fsm[F_LNG + col], bb = fsm[F_LNB + col];
#pragma unroll
      for (int j = 0; j < 8; ++j) {
        float v = (acc[t][j] - mu[j]) * rs[j] * gg + bb;
        stw[(j + 8 * half) * SP + col] = (_Float16)v;
      }
    }
    asm volatile("s_wait_dscnt 0" ::: "memory");

    float z[8];
    {
      float part[8];
#pragma unroll
      for (int e = 0; e < 8; ++e) part[e] = 0.f;
      const _Float16* hrow = stw + l16 * SP;
#pragma unroll 8
      for (int c = 0; c < 32; ++c) {
        const int cc = 32 * half + c;
        const float hv = (float)hrow[cc];
#pragma unroll
        for (int e = 0; e < 8; ++e) part[e] += hv * fsm[F_GW2 + e * 64 + cc];
      }
#pragma unroll
      for (int e = 0; e < 8; ++e)
        z[e] = part[e] + __shfl_xor(part[e], 16, 32) + fsm[F_GB2 + e];
    }
    // top-2 on logits (softmax is monotonic -> identical selection)
    float m1 = -3.0e38f, m2 = -3.0e38f;
#pragma unroll
    for (int e = 0; e < 8; ++e) {
      const float v = z[e];
      if (v > m1) { m2 = m1; m1 = v; } else if (v > m2) { m2 = v; }
    }
    unsigned bits = 0;
#pragma unroll
    for (int e = 0; e < 8; ++e) if (z[e] >= m2) bits |= (1u << e);
    if (half == 0) msw[l16] = bits;
#pragma unroll
    for (int e = 0; e < 8; ++e) {
      unsigned long long bal = __ballot((half == 0) && ((bits >> e) & 1u));
      if (lane == e) cntAcc += (float)__popcll(bal);
    }
    asm volatile("s_wait_dscnt 0" ::: "memory");

    // ---- experts: 3-layer SIREN each, dense over all 8, masked sum ----
    float outAcc = 0.f;
    for (int e = 0; e < NEXP; ++e) {
      const float w0 = 0.5f * BWD + BWD * (float)e;
      zeroAcc(acc);
      gemm64(af, Wlds + (1 + e) * MATSZ, acc, l16, half);
      actStage(acc, fsm + F_BE0 + e * 64, w0, stw, l16, half);
      asm volatile("s_wait_dscnt 0" ::: "memory");
      AFrag ah[2];
      reloadA(stw, ah, l16, half);

      zeroAcc(acc);
      gemm64(ah, Wlds + (9 + e) * MATSZ, acc, l16, half);
      actStage(acc, fsm + F_BM + e * 64, 30.f, stw, l16, half);
      asm volatile("s_wait_dscnt 0" ::: "memory");
      reloadA(stw, ah, l16, half);

      zeroAcc(acc);
      gemm64(ah, Wlds + (17 + e) * MATSZ, acc, l16, half);
      actStage(acc, fsm + F_BM + 512 + e * 64, 30.f, stw, l16, half);
      asm volatile("s_wait_dscnt 0" ::: "memory");

      // x[row,e] = h . wl[e] + bl[e]  (split 32+32 cols across halves)
      float p = 0.f;
      const _Float16* hrow = stw + l16 * SP;
#pragma unroll 8
      for (int c = 0; c < 32; ++c) {
        const int cc = 32 * half + c;
        p += (float)hrow[cc] * fsm[F_WL + e * 64 + cc];
      }
      const float x = p + __shfl_xor(p, 16, 32) + fsm[F_BL + e];
      if ((msw[l16] >> e) & 1u) outAcc += x;
    }
    if (half == 0) out[tile * 16 + l16] = outAcc;
  }
  if (lane < 8) atomicAdd(&counts[lane], cntAcc);
}

__global__ void moe_init(float* counts) {
  if (threadIdx.x < 8) counts[threadIdx.x] = 0.f;
}

__global__ void moe_aux(const float* __restrict__ counts, float* __restrict__ out) {
  if (threadIdx.x == 0) {
    float s = 0.f;
    for (int e = 0; e < 8; ++e) s += counts[e] * counts[e];
    out[N_PTS] = s * (8.0f / ((float)N_PTS * (float)N_PTS));
  }
}

extern "C" void kernel_launch(void* const* d_in, const int* in_sizes, int n_in,
                              void* d_out, int out_size, void* d_ws, size_t ws_size,
                              hipStream_t stream) {
  const float* coords = (const float*)d_in[0];
  const float* fw     = (const float*)d_in[1];
  const float* fb     = (const float*)d_in[2];
  const float* gw1    = (const float*)d_in[3];
  const float* gb1    = (const float*)d_in[4];
  const float* lng    = (const float*)d_in[5];
  const float* lnb    = (const float*)d_in[6];
  const float* gw2    = (const float*)d_in[7];
  const float* gb2    = (const float*)d_in[8];
  const float* we0    = (const float*)d_in[9];
  const float* be0    = (const float*)d_in[10];
  const float* wm     = (const float*)d_in[11];
  const float* bm     = (const float*)d_in[12];
  const float* wl     = (const float*)d_in[13];
  const float* bl     = (const float*)d_in[14];
  float* out    = (float*)d_out;
  float* counts = (float*)d_ws;

  moe_init<<<1, 32, 0, stream>>>(counts);
  moe_kernel<<<BLOCKS, WAVES * 32, SMEM_BYTES, stream>>>(
      coords, fw, fb, gw1, gb1, lng, lnb, gw2, gb2, we0, be0, bm, wm, wl, bl,
      out, counts);
  moe_aux<<<1, 32, 0, stream>>>(counts, out);
}
